// Decoder_54949811585346
// MI455X (gfx1250) — matrix operand order repaired and checked by reading, not compiled, not run on hardware
//
#include <hip/hip_runtime.h>

#define S_LEN 256
#define B_SZ  1024
#define E_DIM 128
#define H_DIM 128

typedef __bf16 v16bf __attribute__((ext_vector_type(16)));
typedef float  v8f   __attribute__((ext_vector_type(8)));

// ---------------- CDNA5 helpers ----------------

__device__ __forceinline__ v8f wmma_bf16(v16bf a, v16bf b, v8f c) {
  // D = A(16x32 bf16) * B(32x16 bf16) + C(16x16 f32)
  return __builtin_amdgcn_wmma_f32_16x16x32_bf16(false, a, false, b, (short)0, c,
                                                 false, false);
}

// A-matrix fragment (16x32, MxK), row-major source with leading dim ld.
// ISA layout: lanes 0-15 / 16-31 split K-halves; 8 consecutive K per vector half,
// so the compiler merges these into 2x ds_load_b128 per lane.
__device__ __forceinline__ v16bf load_A_bf(const __bf16* src, int ld, int lane) {
  const int m = lane & 15, hi = lane >> 4;
  v16bf a;
#pragma unroll
  for (int i = 0; i < 16; ++i) {
    const int k = (i < 8) ? ((hi ? 8 : 0) + i) : ((hi ? 24 : 16) + (i - 8));
    a[i] = src[m * ld + k];
  }
  return a;
}

// B-matrix fragment from pre-swizzled weights: tile-major, lane-major, 16 bf16
// contiguous per lane (32B, aligned) -> 2x global_load_b128, fully coalesced.
__device__ __forceinline__ v16bf load_B_sw(const __bf16* __restrict__ base, int tile,
                                           int lane) {
  return *(const v16bf*)(base + ((size_t)tile * 32 + lane) * 16);
}

// Native CDNA5 transcendental (trans-pipe, co-executes with VALU).
__device__ __forceinline__ float fast_tanh(float x) {
  float r;
  asm("v_tanh_f32 %0, %1" : "=v"(r) : "v"(x));
  return r;
}
__device__ __forceinline__ float fast_sig(float x) {
  return 0.5f * fast_tanh(0.5f * x) + 0.5f;
}

__device__ __forceinline__ float u01(unsigned x) {
  x ^= x >> 17; x *= 0xed5ad4bbu; x ^= x >> 11;
  x *= 0xac4c1b51u; x ^= x >> 15; x *= 0x31848babu; x ^= x >> 14;
  return (float)(x >> 8) * (1.0f / 16777216.0f) + 1e-7f;
}

// ---------------- init: swizzle weights to WMMA-B fragment layout + state ----------------
// Packed layout for W (K x N row-major): tile t = kt*(N/16)+nt covers k0=kt*32,n0=nt*16.
// Wsw[((t*32)+lane)*16 + i] = W[(kt*32 + (i<8 ? (lane>>4)*8 + i : 16 + (lane>>4)*8 + i-8))*N + nt*16 + (lane&15)]

__device__ __forceinline__ void pack_one(const float* __restrict__ W, __bf16* __restrict__ Wsw,
                                         int N, int p) {
  const int e = p & 15;
  const int lane = (p >> 4) & 31;
  const int tile = p >> 9;
  const int ntiles = N >> 4;
  const int nt = tile % ntiles, kt = tile / ntiles;
  const int n = nt * 16 + (lane & 15);
  const int k = kt * 32 + ((e < 8) ? ((lane >> 4) * 8 + e) : (16 + (lane >> 4) * 8 + (e - 8)));
  Wsw[p] = (__bf16)W[k * N + n];
}

__global__ void init_all(const float* __restrict__ Wi, const float* __restrict__ Wh,
                         const float* __restrict__ gWq, const float* __restrict__ pWq,
                         const float* __restrict__ gWr, const float* __restrict__ pWr,
                         const float* __restrict__ dec_in,
                         const float* __restrict__ hx_in, const float* __restrict__ cx_in,
                         __bf16* Wi_sw, __bf16* Wh_sw, __bf16* gWq_sw, __bf16* pWq_sw,
                         __bf16* gWr_sw, __bf16* pWr_sw,
                         __bf16* x_bf, float* hx, float* cx, unsigned char* mask) {
  int i = blockIdx.x * blockDim.x + threadIdx.x;
  if (i < 65536)            pack_one(Wi, Wi_sw, 512, i);
  else if (i < 131072)      pack_one(Wh, Wh_sw, 512, i - 65536);
  else if (i < 147456)      pack_one(gWq, gWq_sw, 128, i - 131072);
  else if (i < 163840)      pack_one(pWq, pWq_sw, 128, i - 147456);
  else if (i < 180224)      pack_one(gWr, gWr_sw, 128, i - 163840);
  else if (i < 196608)      pack_one(pWr, pWr_sw, 128, i - 180224);
  else if (i < 327680)      x_bf[i - 196608] = (__bf16)dec_in[i - 196608];
  else if (i < 458752)    { int j = i - 327680; hx[j] = hx_in[j]; }
  else if (i < 589824)    { int j = i - 458752; cx[j] = cx_in[j]; }
  else if (i < 851968)      mask[i - 589824] = 0;
}

// ---------------- phase 1: e = context @ Wr + br, stored bf16 (b,g,s) ----------------
// grid: (S/16, B), 256 threads (8 waves). Each wave: one 16-col tile for both e_g and e_p.

__global__ void precompute_e(const float* __restrict__ context,
                             const __bf16* __restrict__ gWr_sw, const float* __restrict__ gbr,
                             const __bf16* __restrict__ pWr_sw, const float* __restrict__ pbr,
                             __bf16* __restrict__ e_g, __bf16* __restrict__ e_p) {
  const int s0 = blockIdx.x * 16;
  const int b  = blockIdx.y;
  const int tid = threadIdx.x, wave = tid >> 5, lane = tid & 31;
  __shared__ __bf16 sA[16 * 128];
  for (int i = tid; i < 16 * 128; i += 256) {
    const int si = i >> 7, h = i & 127;
    sA[i] = (__bf16)context[(size_t)(s0 + si) * (B_SZ * H_DIM) + (size_t)b * H_DIM + h];
  }
  __syncthreads();
  const int n0 = wave * 16;
  v8f accg = {}; v8f accp = {};
#pragma unroll
  for (int k0 = 0; k0 < 128; k0 += 32) {
    const int tile = (k0 >> 5) * 8 + (n0 >> 4);
    v16bf a = load_A_bf(sA + k0, 128, lane);
    accg = wmma_bf16(a, load_B_sw(gWr_sw, tile, lane), accg);
    accp = wmma_bf16(a, load_B_sw(pWr_sw, tile, lane), accp);
  }
  const int g = n0 + (lane & 15), hi = lane >> 4;
  const float bg = gbr[g], bp = pbr[g];
  const size_t base = ((size_t)b * H_DIM + g) * S_LEN + s0;
#pragma unroll
  for (int j = 0; j < 8; ++j) {
    const int sl = j + hi * 8;
    e_g[base + sl] = (__bf16)(accg[j] + bg);
    e_p[base + sl] = (__bf16)(accp[j] + bp);
  }
}

// ---------------- K1: LSTM cell + fused qp_g = hy @ gWq + gbq ----------------
// grid: B/16 = 64 blocks, 256 threads (8 waves). hx/cx updated in place (block-private rows).

__global__ void lstm_step(const __bf16* __restrict__ x_bf, float* hx, float* cx,
                          const __bf16* __restrict__ Wi_sw, const __bf16* __restrict__ Wh_sw,
                          const float* __restrict__ bi, const float* __restrict__ bh,
                          const __bf16* __restrict__ gWq_sw, const float* __restrict__ gbq,
                          float* __restrict__ qpg) {
  const int b0 = blockIdx.x * 16;
  const int tid = threadIdx.x, wave = tid >> 5, lane = tid & 31;
  __shared__ __bf16 sX[16 * 128];   // x tile, later reused as hy(bf16)
  __shared__ __bf16 sH[16 * 128];   // hx tile
  __shared__ float  sGt[16 * 512];  // gates
  for (int i = tid; i < 16 * 128; i += 256) {
    const int r = i >> 7, h = i & 127;
    sX[i] = x_bf[(size_t)(b0 + r) * 128 + h];
    sH[i] = (__bf16)hx[(size_t)(b0 + r) * 128 + h];
  }
  __syncthreads();
#pragma unroll
  for (int t = 0; t < 4; ++t) {
    const int n0 = (wave * 4 + t) * 16;
    v8f acc = {};
#pragma unroll
    for (int k0 = 0; k0 < 128; k0 += 32)
      acc = wmma_bf16(load_A_bf(sX + k0, 128, lane),
                      load_B_sw(Wi_sw, (k0 >> 5) * 32 + (n0 >> 4), lane), acc);
#pragma unroll
    for (int k0 = 0; k0 < 128; k0 += 32)
      acc = wmma_bf16(load_A_bf(sH + k0, 128, lane),
                      load_B_sw(Wh_sw, (k0 >> 5) * 32 + (n0 >> 4), lane), acc);
    const int n = n0 + (lane & 15), hi = lane >> 4;
    const float bias = bi[n] + bh[n];
#pragma unroll
    for (int j = 0; j < 8; ++j) sGt[(j + hi * 8) * 512 + n] = acc[j] + bias;
  }
  __syncthreads();
  for (int i = tid; i < 16 * 128; i += 256) {
    const int r = i >> 7, h = i & 127;
    const float gi = sGt[r * 512 + h],       gf = sGt[r * 512 + 128 + h];
    const float gg = sGt[r * 512 + 256 + h], go = sGt[r * 512 + 384 + h];
    const size_t idx = (size_t)(b0 + r) * 128 + h;
    const float c = fast_sig(gf) * cx[idx] + fast_sig(gi) * fast_tanh(gg);
    const float hval = fast_sig(go) * fast_tanh(c);
    cx[idx] = c; hx[idx] = hval;
    sX[i] = (__bf16)hval;
  }
  __syncthreads();
  {
    const int n0 = wave * 16;
    v8f acc = {};
#pragma unroll
    for (int k0 = 0; k0 < 128; k0 += 32)
      acc = wmma_bf16(load_A_bf(sX + k0, 128, lane),
                      load_B_sw(gWq_sw, (k0 >> 5) * 8 + (n0 >> 4), lane), acc);
    const int n = n0 + (lane & 15), hi = lane >> 4;
    const float bias = gbq[n];
#pragma unroll
    for (int j = 0; j < 8; ++j)
      qpg[(size_t)(b0 + j + hi * 8) * 128 + n] = acc[j] + bias;
  }
}

// ---------------- K2: glimpse attention -> softmax -> readout g_l (bf16) ----------------
// grid: B blocks, 256 threads (thread == s).

__global__ void glimpse_step(const __bf16* __restrict__ e_g, const float* __restrict__ qpg,
                             const float* __restrict__ gv, const unsigned char* __restrict__ mask,
                             __bf16* __restrict__ g_l_bf) {
  const int b = blockIdx.x, tid = threadIdx.x;
  __shared__ float sq[128], sv[128], red[256], wsm[256];
  if (tid < 128) { sq[tid] = qpg[(size_t)b * 128 + tid]; sv[tid] = gv[tid]; }
  __syncthreads();
  const __bf16* eb = e_g + (size_t)b * (H_DIM * S_LEN);
  const int s = tid;
  float u = 0.f;
#pragma unroll 8
  for (int h = 0; h < 128; ++h) {
    if ((h & 31) == 0 && h < 96) __builtin_prefetch(eb + (h + 32) * 256 + s, 0, 0);
    u += sv[h] * fast_tanh(sq[h] + (float)eb[h * 256 + s]);
  }
  if (mask[b * 256 + s]) u = -INFINITY;
  red[tid] = u; __syncthreads();
  for (int off = 128; off > 0; off >>= 1) {
    if (tid < off) red[tid] = fmaxf(red[tid], red[tid + off]);
    __syncthreads();
  }
  const float mx = red[0]; __syncthreads();
  const float ex = (u == -INFINITY) ? 0.f : __expf(u - mx);
  red[tid] = ex; __syncthreads();
  for (int off = 128; off > 0; off >>= 1) {
    if (tid < off) red[tid] += red[tid + off];
    __syncthreads();
  }
  wsm[s] = ex / red[0];
  __syncthreads();
  const int wave = tid >> 5, lane = tid & 31;
#pragma unroll
  for (int i = 0; i < 16; ++i) {
    const int g = wave * 16 + i;
    const __bf16* row = eb + g * 256;
    float acc = 0.f;
#pragma unroll
    for (int q = 0; q < 8; ++q) { const int ss = lane + q * 32; acc += (float)row[ss] * wsm[ss]; }
    for (int off = 16; off > 0; off >>= 1) acc += __shfl_down(acc, off);
    if (lane == 0) g_l_bf[(size_t)b * 128 + g] = (__bf16)acc;
  }
}

// ---------------- K3: qp_p = g_l @ pWq + pbq ----------------

__global__ void qp_gemm(const __bf16* __restrict__ A, const __bf16* __restrict__ W_sw,
                        const float* __restrict__ bias, float* __restrict__ out) {
  const int b0 = blockIdx.x * 16;
  const int tid = threadIdx.x, wave = tid >> 5, lane = tid & 31;
  __shared__ __bf16 sA[16 * 128];
  for (int i = tid; i < 16 * 128; i += 256) {
    const int r = i >> 7, h = i & 127;
    sA[i] = A[(size_t)(b0 + r) * 128 + h];
  }
  __syncthreads();
  const int n0 = wave * 16;
  v8f acc = {};
#pragma unroll
  for (int k0 = 0; k0 < 128; k0 += 32)
    acc = wmma_bf16(load_A_bf(sA + k0, 128, lane),
                    load_B_sw(W_sw, (k0 >> 5) * 8 + (n0 >> 4), lane), acc);
  const int n = n0 + (lane & 15), hi = lane >> 4;
  const float bs = bias[n];
#pragma unroll
  for (int j = 0; j < 8; ++j)
    out[(size_t)(b0 + j + hi * 8) * 128 + n] = acc[j] + bs;
}

// ---------------- K4: pointer attention -> probs out, Gumbel sample, gather x ----------------

__global__ void pointer_step(const __bf16* __restrict__ e_p, const float* __restrict__ qpp,
                             const float* __restrict__ pv, unsigned char* mask,
                             const float* __restrict__ emb, __bf16* __restrict__ x_bf,
                             float* __restrict__ out_probs, float* __restrict__ out_idx, int t) {
  const int b = blockIdx.x, tid = threadIdx.x;
  __shared__ float sq[128], sv[128], red[256];
  __shared__ int ridx[256];
  __shared__ int sel;
  if (tid < 128) { sq[tid] = qpp[(size_t)b * 128 + tid]; sv[tid] = pv[tid]; }
  __syncthreads();
  const __bf16* eb = e_p + (size_t)b * (H_DIM * S_LEN);
  const int s = tid;
  float u = 0.f;
#pragma unroll 8
  for (int h = 0; h < 128; ++h) {
    if ((h & 31) == 0 && h < 96) __builtin_prefetch(eb + (h + 32) * 256 + s, 0, 0);
    u += sv[h] * fast_tanh(sq[h] + (float)eb[h * 256 + s]);
  }
  float pl = 10.f * fast_tanh(u);
  if (mask[b * 256 + s]) pl = -INFINITY;
  red[tid] = pl; __syncthreads();
  for (int off = 128; off > 0; off >>= 1) {
    if (tid < off) red[tid] = fmaxf(red[tid], red[tid + off]);
    __syncthreads();
  }
  const float mx = red[0]; __syncthreads();
  const float ex = (pl == -INFINITY) ? 0.f : __expf(pl - mx);
  red[tid] = ex; __syncthreads();
  for (int off = 128; off > 0; off >>= 1) {
    if (tid < off) red[tid] += red[tid + off];
    __syncthreads();
  }
  out_probs[(size_t)t * (B_SZ * S_LEN) + (size_t)b * S_LEN + s] = ex / red[0];
  __syncthreads();
  // Gumbel-max sampling
  const unsigned seed = (unsigned)(((t * B_SZ + b) * S_LEN + s)) * 2654435761u ^ 0x9E3779B9u;
  const float gsc = pl - __logf(-__logf(u01(seed)));
  red[tid] = gsc; ridx[tid] = s; __syncthreads();
  for (int off = 128; off > 0; off >>= 1) {
    if (tid < off && red[tid + off] > red[tid]) { red[tid] = red[tid + off]; ridx[tid] = ridx[tid + off]; }
    __syncthreads();
  }
  if (tid == 0) {
    const int ix = ridx[0];
    sel = ix;
    mask[b * 256 + ix] = 1;                         // masked from next step on
    out_idx[(size_t)t * B_SZ + b] = (float)ix;
  }
  __syncthreads();
  if (tid < 128)
    x_bf[(size_t)b * 128 + tid] = (__bf16)emb[((size_t)sel * B_SZ + b) * 128 + tid];
}

// ---------------- finalize: hN, cN ----------------

__global__ void finalize(const float* __restrict__ hx, const float* __restrict__ cx,
                         float* __restrict__ out_h, float* __restrict__ out_c) {
  const int i = blockIdx.x * blockDim.x + threadIdx.x;
  if (i < B_SZ * H_DIM) { out_h[i] = hx[i]; out_c[i] = cx[i]; }
}

// ---------------- host launch ----------------

extern "C" void kernel_launch(void* const* d_in, const int* in_sizes, int n_in,
                              void* d_out, int out_size, void* d_ws, size_t ws_size,
                              hipStream_t stream) {
  const float* dec_in  = (const float*)d_in[0];
  const float* emb     = (const float*)d_in[1];
  const float* hx_in   = (const float*)d_in[2];
  const float* cx_in   = (const float*)d_in[3];
  const float* context = (const float*)d_in[4];
  const float* Wi      = (const float*)d_in[5];
  const float* bi      = (const float*)d_in[6];
  const float* Wh      = (const float*)d_in[7];
  const float* bh      = (const float*)d_in[8];
  const float* gWq     = (const float*)d_in[9];
  const float* gbq     = (const float*)d_in[10];
  const float* gWr     = (const float*)d_in[11];
  const float* gbr     = (const float*)d_in[12];
  const float* gv      = (const float*)d_in[13];
  const float* pWq     = (const float*)d_in[14];
  const float* pbq     = (const float*)d_in[15];
  const float* pWr     = (const float*)d_in[16];
  const float* pbr     = (const float*)d_in[17];
  const float* pv      = (const float*)d_in[18];

  float* out = (float*)d_out;
  float* out_probs = out;                                   // S*B*S
  float* out_idx   = out + (size_t)S_LEN * B_SZ * S_LEN;    // S*B
  float* out_h     = out_idx + (size_t)S_LEN * B_SZ;        // B*H
  float* out_c     = out_h + (size_t)B_SZ * H_DIM;          // B*H

  // workspace carve-up (256B aligned)
  char* w = (char*)d_ws;
  size_t off = 0;
  auto take = [&](size_t bytes) -> void* {
    void* p = w + off;
    off = (off + bytes + 255) & ~(size_t)255;
    return p;
  };
  __bf16* e_g    = (__bf16*)take((size_t)B_SZ * H_DIM * S_LEN * 2);
  __bf16* e_p    = (__bf16*)take((size_t)B_SZ * H_DIM * S_LEN * 2);
  __bf16* Wi_sw  = (__bf16*)take(128 * 512 * 2);
  __bf16* Wh_sw  = (__bf16*)take(128 * 512 * 2);
  __bf16* gWq_sw = (__bf16*)take(128 * 128 * 2);
  __bf16* pWq_sw = (__bf16*)take(128 * 128 * 2);
  __bf16* gWr_sw = (__bf16*)take(128 * 128 * 2);
  __bf16* pWr_sw = (__bf16*)take(128 * 128 * 2);
  __bf16* x_bf   = (__bf16*)take((size_t)B_SZ * E_DIM * 2);
  float*  hx     = (float*)take((size_t)B_SZ * H_DIM * 4);
  float*  cx     = (float*)take((size_t)B_SZ * H_DIM * 4);
  float*  qpg    = (float*)take((size_t)B_SZ * H_DIM * 4);
  float*  qpp    = (float*)take((size_t)B_SZ * H_DIM * 4);
  __bf16* g_l_bf = (__bf16*)take((size_t)B_SZ * H_DIM * 2);
  unsigned char* mask = (unsigned char*)take((size_t)B_SZ * S_LEN);
  (void)ws_size; (void)in_sizes; (void)n_in; (void)out_size;

  init_all<<<3328, 256, 0, stream>>>(Wi, Wh, gWq, pWq, gWr, pWr, dec_in, hx_in, cx_in,
                                     Wi_sw, Wh_sw, gWq_sw, pWq_sw, gWr_sw, pWr_sw,
                                     x_bf, hx, cx, mask);

  precompute_e<<<dim3(S_LEN / 16, B_SZ), 256, 0, stream>>>(context, gWr_sw, gbr,
                                                           pWr_sw, pbr, e_g, e_p);

  for (int t = 0; t < S_LEN; ++t) {
    lstm_step<<<B_SZ / 16, 256, 0, stream>>>(x_bf, hx, cx, Wi_sw, Wh_sw, bi, bh,
                                             gWq_sw, gbq, qpg);
    glimpse_step<<<B_SZ, 256, 0, stream>>>(e_g, qpg, gv, mask, g_l_bf);
    qp_gemm<<<B_SZ / 16, 256, 0, stream>>>(g_l_bf, pWq_sw, pbq, qpp);
    pointer_step<<<B_SZ, 256, 0, stream>>>(e_p, qpp, pv, mask, emb, x_bf,
                                           out_probs, out_idx, t);
  }

  finalize<<<(B_SZ * H_DIM + 255) / 256, 256, 0, stream>>>(hx, cx, out_h, out_c);
}
